// RWKV_TimeMix_59674275611175
// MI455X (gfx1250) — matrix operand order repaired and checked by reading, not compiled, not run on hardware
//
#include <hip/hip_runtime.h>
#include <hip/hip_bf16.h>

typedef __attribute__((ext_vector_type(16))) __bf16 v16bf;
typedef __attribute__((ext_vector_type(8)))  __bf16 v8bf;
typedef __attribute__((ext_vector_type(8)))  float  v8f;

// ---------------------------------------------------------------------------
// 1) Time-shift mixing: xk/xv/xr = x*mix + x[t-1]*(1-mix), cast to bf16.
// ---------------------------------------------------------------------------
__global__ __launch_bounds__(256)
void prep_timemix(const float* __restrict__ x,
                  const float* __restrict__ mk,
                  const float* __restrict__ mv,
                  const float* __restrict__ mr,
                  __bf16* __restrict__ xk,
                  __bf16* __restrict__ xv,
                  __bf16* __restrict__ xr,
                  int T, int C, size_t total)
{
    size_t e = (size_t)blockIdx.x * 256 + threadIdx.x;
    if (e >= total) return;
    int c = (int)(e % C);
    int t = (int)((e / C) % T);
    float xc = x[e];
    float xp = (t > 0) ? x[e - C] : 0.0f;
    float a = mk[c];
    float b = mv[c];
    float r = mr[c];
    xk[e] = (__bf16)(xc * a + xp * (1.0f - a));
    xv[e] = (__bf16)(xc * b + xp * (1.0f - b));
    xr[e] = (__bf16)(xc * r + xp * (1.0f - r));
}

// ---------------------------------------------------------------------------
// 2) Weight transpose + bf16 convert: WT[k*C + n] = (bf16) W[n*C + k].
//    blockIdx.y selects which of the 4 weights.
// ---------------------------------------------------------------------------
__global__ __launch_bounds__(256)
void transpose_w(const float* __restrict__ W0, const float* __restrict__ W1,
                 const float* __restrict__ W2, const float* __restrict__ W3,
                 __bf16* __restrict__ T0, __bf16* __restrict__ T1,
                 __bf16* __restrict__ T2, __bf16* __restrict__ T3,
                 int C)
{
    int idx = blockIdx.x * 256 + threadIdx.x;           // idx = n*C + k
    if (idx >= C * C) return;
    const float* src; __bf16* dst;
    switch (blockIdx.y) {
        case 0:  src = W0; dst = T0; break;
        case 1:  src = W1; dst = T1; break;
        case 2:  src = W2; dst = T2; break;
        default: src = W3; dst = T3; break;
    }
    int n = idx / C;
    int k = idx % C;
    dst[(size_t)k * C + n] = (__bf16)src[idx];
}

// ---------------------------------------------------------------------------
// 3) bf16 WMMA GEMM:  Out[m, n] = sum_k A[m, k] * BT[k, n]   (f32 accumulate)
//    Block = 256 threads = 8 waves; block tile 128(M) x 128(N).
//    Waves arranged 4(M) x 2(N); each wave owns 32x64 = 2x4 WMMA tiles.
//    Fragment layouts per CDNA5 ISA 7.12.2 (wave32):
//      A 16x32 bf16: lane L -> m = L%16, half = L/16;
//                    v[0..7]  = K = half*8 + 0..7
//                    v[8..15] = K = 16 + half*8 + 0..7
//      B 32x16 bf16: lane L -> n = L%16, half = L/16; v[] K order same as A
//      C/D 16x16 f32: vgpr r -> M = r + 8*half, N = lane%16
// ---------------------------------------------------------------------------
__global__ __launch_bounds__(256)
void gemm_bf16(const __bf16* __restrict__ A,   // [M, K] bf16 row-major
               const __bf16* __restrict__ BT,  // [K, N] bf16 row-major (= W^T)
               float* __restrict__ Out,        // [M, N] f32
               int N, int K)
{
    const int lane = threadIdx.x & 31;
    const int wave = threadIdx.x >> 5;
    const int wm   = wave & 3;          // 0..3 -> 32-row slice
    const int wn   = wave >> 2;         // 0..1 -> 64-col slice
    const int half = lane >> 4;         // 0 or 1
    const int mrow = lane & 15;

    const int m_base = blockIdx.x * 128 + wm * 32;
    const int n_base = blockIdx.y * 128 + wn * 64;

    v8f acc[2][4] = {};

    for (int k0 = 0; k0 < K; k0 += 32) {
        // --- A fragments (two 16x32 tiles stacked in M) ---
        v16bf afrag[2];
#pragma unroll
        for (int i = 0; i < 2; ++i) {
            const __bf16* arow =
                A + (size_t)(m_base + i * 16 + mrow) * K + k0 + half * 8;
            v8bf lo = *(const v8bf*)(arow);        // K = half*8 + 0..7
            v8bf hi = *(const v8bf*)(arow + 16);   // K = 16 + half*8 + 0..7
#pragma unroll
            for (int e = 0; e < 8; ++e) {
                afrag[i][e]     = lo[e];
                afrag[i][8 + e] = hi[e];
            }
        }
        // --- B fragments (four 32x16 tiles side by side in N) ---
        const __bf16* brow = BT + (size_t)(k0 + half * 8) * N + n_base + mrow;
        v16bf bfrag[4];
#pragma unroll
        for (int j = 0; j < 4; ++j)
#pragma unroll
            for (int e = 0; e < 8; ++e) {
                bfrag[j][e]     = brow[(size_t)e * N + j * 16];
                bfrag[j][8 + e] = brow[(size_t)(16 + e) * N + j * 16];
            }

        // --- 8 WMMAs ---
#pragma unroll
        for (int i = 0; i < 2; ++i)
#pragma unroll
            for (int j = 0; j < 4; ++j)
                acc[i][j] = __builtin_amdgcn_wmma_f32_16x16x32_bf16(
                    false, afrag[i], false, bfrag[j],
                    (short)0, acc[i][j], false, false);
    }

    // --- store D (f32) ---
#pragma unroll
    for (int i = 0; i < 2; ++i) {
#pragma unroll
        for (int j = 0; j < 4; ++j) {
#pragma unroll
            for (int r = 0; r < 8; ++r) {
                int m = m_base + i * 16 + r + 8 * half;
                int n = n_base + j * 16 + (lane & 15);
                Out[(size_t)m * N + n] = acc[i][j][r];
            }
        }
    }
}

// ---------------------------------------------------------------------------
// 4) WKV scan (log-domain, numerically stable) + sigmoid(r) gate, bf16 out.
//    One thread per (b, c) channel; sequential over T; coalesced in c.
// ---------------------------------------------------------------------------
__global__ __launch_bounds__(256)
void wkv_scan(const float* __restrict__ k,
              const float* __restrict__ v,
              const float* __restrict__ r,
              const float* __restrict__ time_decay,
              const float* __restrict__ time_first,
              __bf16* __restrict__ rwkv,
              int T, int C, int BC)
{
    int tid = blockIdx.x * 256 + threadIdx.x;
    if (tid >= BC) return;
    int b = tid / C;
    int c = tid % C;

    const float w = -__expf(time_decay[c]);
    const float u = time_first[c];

    float aa = 0.0f, bb = 0.0f, pp = -1e38f;
    size_t base = (size_t)b * T * C + c;

    for (int t = 0; t < T; ++t) {
        size_t idx = base + (size_t)t * C;
        float kt = k[idx];
        float vt = v[idx];

        float ww = u + kt;
        float p  = fmaxf(pp, ww);
        float e1 = __expf(pp - p);
        float e2 = __expf(ww - p);
        float y  = (e1 * aa + e2 * vt) / (e1 * bb + e2);

        float rs = 1.0f / (1.0f + __expf(-r[idx]));
        rwkv[idx] = (__bf16)(y * rs);

        float ww2 = pp + w;
        float p2  = fmaxf(ww2, kt);
        float e1b = __expf(ww2 - p2);
        float e2b = __expf(kt - p2);
        aa = e1b * aa + e2b * vt;
        bb = e1b * bb + e2b;
        pp = p2;
    }
}

// ---------------------------------------------------------------------------
// Launch
// ---------------------------------------------------------------------------
extern "C" void kernel_launch(void* const* d_in, const int* in_sizes, int n_in,
                              void* d_out, int out_size, void* d_ws, size_t ws_size,
                              hipStream_t stream)
{
    const float* x  = (const float*)d_in[0];
    const float* td = (const float*)d_in[1];
    const float* tf = (const float*)d_in[2];
    const float* mk = (const float*)d_in[3];
    const float* mv = (const float*)d_in[4];
    const float* mr = (const float*)d_in[5];
    const float* Wk = (const float*)d_in[6];
    const float* Wv = (const float*)d_in[7];
    const float* Wr = (const float*)d_in[8];
    const float* Wo = (const float*)d_in[9];

    const int C  = in_sizes[1];          // 1024
    const int Bn = 32;
    const int T  = in_sizes[0] / (Bn * C);
    const int M  = Bn * T;               // 32768
    const size_t MC = (size_t)M * C;

    // workspace layout
    char* ws = (char*)d_ws;
    size_t o = 0;
    __bf16* xk  = (__bf16*)(ws + o); o += MC * 2;
    __bf16* xv  = (__bf16*)(ws + o); o += MC * 2;
    __bf16* xr  = (__bf16*)(ws + o); o += MC * 2;
    __bf16* WkT = (__bf16*)(ws + o); o += (size_t)C * C * 2;
    __bf16* WvT = (__bf16*)(ws + o); o += (size_t)C * C * 2;
    __bf16* WrT = (__bf16*)(ws + o); o += (size_t)C * C * 2;
    __bf16* WoT = (__bf16*)(ws + o); o += (size_t)C * C * 2;
    float*  kb  = (float*)(ws + o);  o += MC * 4;
    float*  vb  = (float*)(ws + o);  o += MC * 4;
    float*  rb  = (float*)d_out;        // stage r in d_out (overwritten later)
    __bf16* rwkv = xk;                  // alias: xk is dead after GEMM #1

    dim3 blk(256);

    // 1) time-shift mix -> bf16
    prep_timemix<<<dim3((unsigned)(MC / 256)), blk, 0, stream>>>(
        x, mk, mv, mr, xk, xv, xr, T, C, MC);

    // 2) transpose weights -> bf16
    transpose_w<<<dim3((unsigned)((C * C + 255) / 256), 4), blk, 0, stream>>>(
        Wk, Wv, Wr, Wo, WkT, WvT, WrT, WoT, C);

    // 3) k / v / r projections
    dim3 g(M / 128, C / 128);
    gemm_bf16<<<g, blk, 0, stream>>>(xk, WkT, kb, C, C);
    gemm_bf16<<<g, blk, 0, stream>>>(xv, WvT, vb, C, C);
    gemm_bf16<<<g, blk, 0, stream>>>(xr, WrT, rb, C, C);

    // 4) WKV scan + sigmoid gate -> bf16 rwkv
    wkv_scan<<<dim3((unsigned)((Bn * C + 255) / 256)), blk, 0, stream>>>(
        kb, vb, rb, td, tf, rwkv, T, C, Bn * C);

    // 5) output projection
    gemm_bf16<<<g, blk, 0, stream>>>(rwkv, WoT, (float*)d_out, C, C);
}
